// EncodeProcessDecode_40106404610147
// MI455X (gfx1250) — compile-verified, matrix operand order checked by reading, not run on hardware
//
#include <hip/hip_runtime.h>
#include <math.h>

// ---------------------------------------------------------------------------
// Types for CDNA5 WMMA (wave32, v_wmma_f32_16x16x32_bf16)
// ---------------------------------------------------------------------------
typedef unsigned short u16;
typedef __attribute__((ext_vector_type(8)))  unsigned short u16x8;
typedef __attribute__((ext_vector_type(16))) __bf16         v16bf;
typedef __attribute__((ext_vector_type(8)))  float          v8f;

union ABFrag {
    struct { u16x8 lo, hi; } s;
    v16bf v;
};

// f32 -> bf16 with round-to-nearest-even
__device__ __forceinline__ u16 f2bf(float f) {
    union { float f; unsigned u; } c; c.f = f;
    unsigned u = c.u;
    unsigned r = u + 0x7FFFu + ((u >> 16) & 1u);
    return (u16)(r >> 16);
}

#define LN_EPS 1e-5f

// ---------------------------------------------------------------------------
// Fused 2-layer MLP (Linear->ReLU->Linear->ReLU) via bf16 WMMA, f32 accum.
// Gathers/concats up to 3 input blocks of 128 f32 columns (or reads a
// pre-packed bf16 X for the narrow encoders). Writes pre-LayerNorm output
// and accumulates graph-wide sum / sum-of-squares for the LN that follows.
//
// Block = 64 threads (2 waves); each wave owns a 16-row WMMA tile; the block
// processes 32 rows. Weights are pre-transposed to bf16 [128][KIN] so each
// lane's B-fragment is two contiguous 16B loads (L2-resident, shared by all
// blocks).
// ---------------------------------------------------------------------------
template <int KC>  // KIN = KC*32
__global__ __launch_bounds__(64)
void mlp2_wmma_kernel(int rows,
                      const u16* __restrict__ Xsmall,                 // optional packed bf16 [rows][KIN]
                      const float* __restrict__ s0, const int* __restrict__ i0,
                      const float* __restrict__ s1, const int* __restrict__ i1,
                      const float* __restrict__ s2, const int* __restrict__ i2,
                      const u16* __restrict__ W1t, const float* __restrict__ b1,   // W1t: bf16 [128][KIN]
                      const u16* __restrict__ W2t, const float* __restrict__ b2,   // W2t: bf16 [128][128]
                      float* __restrict__ outPre,                     // [rows][128]
                      float* __restrict__ stats)                      // stats[0]=sum, stats[1]=sumsq
{
    constexpr int KIN = KC * 32;
    __shared__ alignas(16) u16 Xs[32 * KIN];
    __shared__ alignas(16) u16 Hs[32 * 128];
    __shared__ float redS[64];
    __shared__ float redQ[64];

    const int tid      = threadIdx.x;
    const int lane     = tid & 31;
    const int wave     = tid >> 5;
    const int blockRow = blockIdx.x * 32;

    // Keep the weight stream hot (global_prefetch_b8).
    __builtin_prefetch(&W1t[(size_t)(tid & 127) * KIN], 0, 0);
    __builtin_prefetch(&W2t[(size_t)(tid & 127) * 128], 0, 0);

    // ---- Stage gathered/concatenated input rows into LDS as bf16 ----------
    for (int i = tid; i < 32 * KIN; i += 64) {
        int r = i / KIN, c = i % KIN;
        int gr = blockRow + r;
        u16 v = 0;
        if (gr < rows) {
            if (Xsmall) {
                v = Xsmall[(size_t)gr * KIN + c];
            } else {
                int blk = c >> 7, cc = c & 127;
                const float* sp = (blk == 0) ? s0 : (blk == 1) ? s1 : s2;
                const int*   ip = (blk == 0) ? i0 : (blk == 1) ? i1 : i2;
                int rr = ip ? ip[gr] : gr;
                v = f2bf(sp[(size_t)rr * 128 + cc]);
            }
        }
        Xs[i] = v;
    }
    __syncthreads();

    // Lane mapping for 16-bit A-fragments (ISA 7.12.2):
    //   lane&15 -> matrix row, (lane>>4)*8 -> K half-offset, chunks at +0/+16.
    const int m16     = lane & 15;
    const int kh      = (lane >> 4) * 8;
    const int rowBase = wave * 16;

    const v8f zero8 = {0.f, 0.f, 0.f, 0.f, 0.f, 0.f, 0.f, 0.f};
    v8f acc[8];
#pragma unroll
    for (int nt = 0; nt < 8; ++nt) acc[nt] = zero8;

    // ---- Layer 1: [16 x KIN] @ [KIN x 128] ---------------------------------
#pragma unroll
    for (int kt = 0; kt < KC; ++kt) {
        ABFrag a;
        const u16* xr = &Xs[(rowBase + m16) * KIN + kt * 32 + kh];
        a.s.lo = *reinterpret_cast<const u16x8*>(xr);
        a.s.hi = *reinterpret_cast<const u16x8*>(xr + 16);
#pragma unroll
        for (int nt = 0; nt < 8; ++nt) {
            ABFrag b;
            const u16* wr = &W1t[(size_t)(nt * 16 + m16) * KIN + kt * 32 + kh];
            b.s.lo = *reinterpret_cast<const u16x8*>(wr);
            b.s.hi = *reinterpret_cast<const u16x8*>(wr + 16);
            acc[nt] = __builtin_amdgcn_wmma_f32_16x16x32_bf16(
                false, a.v, false, b.v, (short)0, acc[nt], false, false);
        }
    }

    // bias + ReLU -> bf16 hidden in LDS (C/D layout: lane holds col m16, rows (lane>>4)*8+j)
#pragma unroll
    for (int nt = 0; nt < 8; ++nt) {
        int n = nt * 16 + m16;
        float bb = b1[n];
#pragma unroll
        for (int j = 0; j < 8; ++j) {
            int m = (lane >> 4) * 8 + j;
            float v = fmaxf(acc[nt][j] + bb, 0.f);
            Hs[(rowBase + m) * 128 + n] = f2bf(v);
        }
    }
    __syncthreads();

    // ---- Layer 2: [16 x 128] @ [128 x 128] ---------------------------------
#pragma unroll
    for (int nt = 0; nt < 8; ++nt) acc[nt] = zero8;
#pragma unroll
    for (int kt = 0; kt < 4; ++kt) {
        ABFrag a;
        const u16* xr = &Hs[(rowBase + m16) * 128 + kt * 32 + kh];
        a.s.lo = *reinterpret_cast<const u16x8*>(xr);
        a.s.hi = *reinterpret_cast<const u16x8*>(xr + 16);
#pragma unroll
        for (int nt = 0; nt < 8; ++nt) {
            ABFrag b;
            const u16* wr = &W2t[(size_t)(nt * 16 + m16) * 128 + kt * 32 + kh];
            b.s.lo = *reinterpret_cast<const u16x8*>(wr);
            b.s.hi = *reinterpret_cast<const u16x8*>(wr + 16);
            acc[nt] = __builtin_amdgcn_wmma_f32_16x16x32_bf16(
                false, a.v, false, b.v, (short)0, acc[nt], false, false);
        }
    }

    // bias + ReLU -> pre-LN output, plus LN statistics
    float sum = 0.f, sq = 0.f;
#pragma unroll
    for (int nt = 0; nt < 8; ++nt) {
        int n = nt * 16 + m16;
        float bb = b2[n];
#pragma unroll
        for (int j = 0; j < 8; ++j) {
            int m  = (lane >> 4) * 8 + j;
            int gr = blockRow + rowBase + m;
            float v = fmaxf(acc[nt][j] + bb, 0.f);
            if (gr < rows) {
                outPre[(size_t)gr * 128 + n] = v;
                sum += v;
                sq  += v * v;
            }
        }
    }
    redS[tid] = sum;
    redQ[tid] = sq;
    __syncthreads();
    for (int s = 32; s > 0; s >>= 1) {
        if (tid < s) { redS[tid] += redS[tid + s]; redQ[tid] += redQ[tid + s]; }
        __syncthreads();
    }
    if (tid == 0) {
        atomicAdd(&stats[0], redS[0]);
        atomicAdd(&stats[1], redQ[0]);
    }
}

// ---------------------------------------------------------------------------
// Graph-mode LayerNorm apply: (x - mean) / (std + eps) * g + beta [+ residual]
// ---------------------------------------------------------------------------
__global__ void ln_apply_kernel(long long total, const float* __restrict__ pre,
                                const float* __restrict__ stats, float cntInv,
                                const float* __restrict__ g, const float* __restrict__ bta,
                                const float* __restrict__ resid, float* __restrict__ out)
{
    float mean = stats[0] * cntInv;
    float var  = fmaxf(stats[1] * cntInv - mean * mean, 0.f);
    float inv  = 1.f / (sqrtf(var) + LN_EPS);
    long long i  = (long long)blockIdx.x * blockDim.x + threadIdx.x;
    long long st = (long long)gridDim.x * blockDim.x;
    for (; i < total; i += st) {
        int c = (int)(i & 127);
        float v = (pre[i] - mean) * inv * g[c] + bta[c];
        if (resid) v += resid[i];
        out[i] = v;
    }
}

// ---------------------------------------------------------------------------
// segment_sum via f32 atomics: aggr[dst[e], c] += msg[e, c]
// ---------------------------------------------------------------------------
__global__ void scatter_add_kernel(const float* __restrict__ msg, const int* __restrict__ dst,
                                   long long total, float* __restrict__ aggr)
{
    long long i  = (long long)blockIdx.x * blockDim.x + threadIdx.x;
    long long st = (long long)gridDim.x * blockDim.x;
    for (; i < total; i += st) {
        int e = (int)(i >> 7), c = (int)(i & 127);
        atomicAdd(&aggr[(size_t)dst[e] * 128 + c], msg[i]);
    }
}

__global__ void zero_f32_kernel(float* __restrict__ p, long long n)
{
    long long i  = (long long)blockIdx.x * blockDim.x + threadIdx.x;
    long long st = (long long)gridDim.x * blockDim.x;
    for (; i < n; i += st) p[i] = 0.f;
}

// ---------------------------------------------------------------------------
// Pack weights: f32 [Kin][128] row-major -> bf16 [128][KinPad] "B-major"
// (zero padded K) so WMMA B-fragments are contiguous per lane.
// ---------------------------------------------------------------------------
__global__ void pack_wt_kernel(const float* __restrict__ W, int Kin, int KinPad,
                               u16* __restrict__ Wt)
{
    int total = 128 * KinPad;
    int i  = blockIdx.x * blockDim.x + threadIdx.x;
    int st = gridDim.x * blockDim.x;
    for (; i < total; i += st) {
        int n = i / KinPad, k = i % KinPad;
        Wt[i] = (k < Kin) ? f2bf(W[k * 128 + n]) : (u16)0;
    }
}

// ---------------------------------------------------------------------------
// Encoder input prep: scaled concat, bf16, K padded to 32.
// ---------------------------------------------------------------------------
__global__ void prep_nodes_kernel(int N, const float* __restrict__ ms, const float* __restrict__ pos,
                                  const float* __restrict__ ntp,
                                  const float* __restrict__ mp, const float* __restrict__ sp,
                                  const float* __restrict__ mms, const float* __restrict__ sms,
                                  u16* __restrict__ Xn)
{
    int i = blockIdx.x * blockDim.x + threadIdx.x;
    if (i >= N) return;
    u16* row = Xn + (size_t)i * 32;
    row[0] = f2bf((ms[i] - mms[0]) / sms[0]);
    row[1] = f2bf((pos[2 * i + 0] - mp[0]) / sp[0]);
    row[2] = f2bf((pos[2 * i + 1] - mp[1]) / sp[1]);
    row[3] = f2bf(ntp[i]);
    for (int c = 4; c < 32; ++c) row[c] = 0;
}

__global__ void prep_edges_kernel(int E, const float* __restrict__ ea,
                                  const float* __restrict__ mew, const float* __restrict__ sew,
                                  u16* __restrict__ Xe)
{
    int i = blockIdx.x * blockDim.x + threadIdx.x;
    if (i >= E) return;
    u16* row = Xe + (size_t)i * 32;
    row[0] = f2bf((ea[i] - mew[0]) / sew[0]);
    for (int c = 1; c < 32; ++c) row[c] = 0;
}

// ---------------------------------------------------------------------------
// Decoder: relu(h@W1+b1)@W2+b2, scaled. Tiny (3 outputs) -> VALU.
// ---------------------------------------------------------------------------
__global__ __launch_bounds__(128)
void decoder_kernel(int N, const float* __restrict__ h,
                    const float* __restrict__ W1, const float* __restrict__ b1,
                    const float* __restrict__ W2, const float* __restrict__ b2,
                    const float* __restrict__ mls, const float* __restrict__ sls,
                    float* __restrict__ out)
{
    __shared__ float hr[128];
    __shared__ float hid[128];
    for (int node = blockIdx.x; node < N; node += gridDim.x) {
        __syncthreads();
        hr[threadIdx.x] = h[(size_t)node * 128 + threadIdx.x];
        __syncthreads();
        float a = b1[threadIdx.x];
        for (int k = 0; k < 128; ++k) a += hr[k] * W1[k * 128 + threadIdx.x];
        hid[threadIdx.x] = fmaxf(a, 0.f);
        __syncthreads();
        if (threadIdx.x < 3) {
            float o = b2[threadIdx.x];
            for (int k = 0; k < 128; ++k) o += hid[k] * W2[k * 3 + threadIdx.x];
            out[(size_t)node * 3 + threadIdx.x] = o * sls[threadIdx.x] + mls[threadIdx.x];
        }
    }
}

// ---------------------------------------------------------------------------
// Host orchestration
// ---------------------------------------------------------------------------
extern "C" void kernel_launch(void* const* d_in, const int* in_sizes, int n_in,
                              void* d_out, int out_size, void* d_ws, size_t ws_size,
                              hipStream_t stream)
{
    const int N = in_sizes[0];          // mean_stress [N]
    const int E = in_sizes[3];          // edge_attr   [E]

    // ---- inputs ------------------------------------------------------------
    const float* mean_stress = (const float*)d_in[0];
    const float* pos         = (const float*)d_in[1];
    const float* nodes_types = (const float*)d_in[2];
    const float* edge_attr   = (const float*)d_in[3];
    const int*   edge_index  = (const int*)d_in[4];
    const float* mean_pos    = (const float*)d_in[5];
    const float* std_pos     = (const float*)d_in[6];
    const float* mean_ms     = (const float*)d_in[7];
    const float* std_ms      = (const float*)d_in[8];
    const float* mean_ls     = (const float*)d_in[9];
    const float* std_ls      = (const float*)d_in[10];
    const float* mean_ew     = (const float*)d_in[11];
    const float* std_ew      = (const float*)d_in[12];

    const float* ne_W1 = (const float*)d_in[13]; const float* ne_b1 = (const float*)d_in[14];
    const float* ne_W2 = (const float*)d_in[15]; const float* ne_b2 = (const float*)d_in[16];
    const float* ne_g  = (const float*)d_in[17]; const float* ne_b  = (const float*)d_in[18];
    const float* ee_W1 = (const float*)d_in[19]; const float* ee_b1 = (const float*)d_in[20];
    const float* ee_W2 = (const float*)d_in[21]; const float* ee_b2 = (const float*)d_in[22];
    const float* ee_g  = (const float*)d_in[23]; const float* ee_b  = (const float*)d_in[24];
    const float* pe_W1 = (const float*)d_in[25]; const float* pe_b1 = (const float*)d_in[26];
    const float* pe_W2 = (const float*)d_in[27]; const float* pe_b2 = (const float*)d_in[28];
    const float* pe_g  = (const float*)d_in[29]; const float* pe_b  = (const float*)d_in[30];
    const float* pn_W1 = (const float*)d_in[31]; const float* pn_b1 = (const float*)d_in[32];
    const float* pn_W2 = (const float*)d_in[33]; const float* pn_b2 = (const float*)d_in[34];
    const float* pn_g  = (const float*)d_in[35]; const float* pn_b  = (const float*)d_in[36];
    const float* dec_W1 = (const float*)d_in[37]; const float* dec_b1 = (const float*)d_in[38];
    const float* dec_W2 = (const float*)d_in[39]; const float* dec_b2 = (const float*)d_in[40];

    const int* srcIdx = edge_index;        // row (source)
    const int* dstIdx = edge_index + E;    // col (target)

    // ---- workspace layout --------------------------------------------------
    size_t off = 0;
    auto ALLOC = [&](size_t bytes) -> void* {
        void* p = (char*)d_ws + off;
        off += (bytes + 255) & ~(size_t)255;
        return p;
    };
    u16* neW1t = (u16*)ALLOC((size_t)128 * 32  * 2);
    u16* neW2t = (u16*)ALLOC((size_t)128 * 128 * 2);
    u16* eeW1t = (u16*)ALLOC((size_t)128 * 32  * 2);
    u16* eeW2t = (u16*)ALLOC((size_t)128 * 128 * 2);
    u16* peW1t = (u16*)ALLOC((size_t)128 * 384 * 2);
    u16* peW2t = (u16*)ALLOC((size_t)128 * 128 * 2);
    u16* pnW1t = (u16*)ALLOC((size_t)128 * 256 * 2);
    u16* pnW2t = (u16*)ALLOC((size_t)128 * 128 * 2);
    float* stats = (float*)ALLOC(16 * sizeof(float));
    u16* Xn = (u16*)ALLOC((size_t)N * 32 * 2);
    u16* Xe = (u16*)ALLOC((size_t)E * 32 * 2);
    float* h0   = (float*)ALLOC((size_t)N * 128 * 4);
    float* h1   = (float*)ALLOC((size_t)N * 128 * 4);
    float* aggr = (float*)ALLOC((size_t)N * 128 * 4);
    float* e0   = (float*)ALLOC((size_t)E * 128 * 4);
    float* e1   = (float*)ALLOC((size_t)E * 128 * 4);
    float* msg  = (float*)ALLOC((size_t)E * 128 * 4);
    float* tmp  = (float*)ALLOC((size_t)E * 128 * 4);
    (void)ws_size; (void)n_in; (void)out_size;

    float* outp = (float*)d_out;

    const int gridN = (N + 31) / 32;
    const int gridE = (E + 31) / 32;
    const long long totN = (long long)N * 128;
    const long long totE = (long long)E * 128;
    const float cntInvN = 1.f / (float)totN;
    const float cntInvE = 1.f / (float)totE;

    auto zero2 = [&]() { zero_f32_kernel<<<1, 64, 0, stream>>>(stats, 2); };

    // ---- pack weights to bf16 B-major --------------------------------------
    pack_wt_kernel<<<64, 256, 0, stream>>>(ne_W1, 4,   32,  neW1t);
    pack_wt_kernel<<<64, 256, 0, stream>>>(ne_W2, 128, 128, neW2t);
    pack_wt_kernel<<<64, 256, 0, stream>>>(ee_W1, 1,   32,  eeW1t);
    pack_wt_kernel<<<64, 256, 0, stream>>>(ee_W2, 128, 128, eeW2t);
    pack_wt_kernel<<<192, 256, 0, stream>>>(pe_W1, 384, 384, peW1t);
    pack_wt_kernel<<<64, 256, 0, stream>>>(pe_W2, 128, 128, peW2t);
    pack_wt_kernel<<<128, 256, 0, stream>>>(pn_W1, 256, 256, pnW1t);
    pack_wt_kernel<<<64, 256, 0, stream>>>(pn_W2, 128, 128, pnW2t);

    // ---- encoders ----------------------------------------------------------
    prep_nodes_kernel<<<(N + 255) / 256, 256, 0, stream>>>(
        N, mean_stress, pos, nodes_types, mean_pos, std_pos, mean_ms, std_ms, Xn);
    prep_edges_kernel<<<(E + 255) / 256, 256, 0, stream>>>(E, edge_attr, mean_ew, std_ew, Xe);

    zero2();
    mlp2_wmma_kernel<1><<<gridN, 64, 0, stream>>>(
        N, Xn, nullptr, nullptr, nullptr, nullptr, nullptr, nullptr,
        neW1t, ne_b1, neW2t, ne_b2, tmp, stats);
    ln_apply_kernel<<<2048, 256, 0, stream>>>(totN, tmp, stats, cntInvN, ne_g, ne_b, nullptr, h0);

    zero2();
    mlp2_wmma_kernel<1><<<gridE, 64, 0, stream>>>(
        E, Xe, nullptr, nullptr, nullptr, nullptr, nullptr, nullptr,
        eeW1t, ee_b1, eeW2t, ee_b2, tmp, stats);
    ln_apply_kernel<<<2048, 256, 0, stream>>>(totE, tmp, stats, cntInvE, ee_g, ee_b, nullptr, e0);

    // ---- processor: 3 message-passing steps --------------------------------
    float* hc = h0; float* hn = h1;
    float* ec = e0; float* en = e1;
    for (int s = 0; s < 3; ++s) {
        // msg = LN(edge_net([h[dst], h[src], e]))
        zero2();
        mlp2_wmma_kernel<12><<<gridE, 64, 0, stream>>>(
            E, nullptr, hc, dstIdx, hc, srcIdx, ec, nullptr,
            peW1t, pe_b1, peW2t, pe_b2, tmp, stats);
        ln_apply_kernel<<<2048, 256, 0, stream>>>(totE, tmp, stats, cntInvE, pe_g, pe_b, nullptr, msg);

        // aggr = segment_sum(msg, dst)
        zero_f32_kernel<<<2048, 256, 0, stream>>>(aggr, totN);
        scatter_add_kernel<<<2048, 256, 0, stream>>>(msg, dstIdx, totE, aggr);

        // h_new = LN(node_net([aggr, h])) + h
        zero2();
        mlp2_wmma_kernel<8><<<gridN, 64, 0, stream>>>(
            N, nullptr, aggr, nullptr, hc, nullptr, nullptr, nullptr,
            pnW1t, pn_b1, pnW2t, pn_b2, tmp, stats);
        ln_apply_kernel<<<2048, 256, 0, stream>>>(totN, tmp, stats, cntInvN, pn_g, pn_b, hc, hn);

        // e_new = LN(edge_net([h[src], h[dst], e])) + e   (uses OLD h)
        zero2();
        mlp2_wmma_kernel<12><<<gridE, 64, 0, stream>>>(
            E, nullptr, hc, srcIdx, hc, dstIdx, ec, nullptr,
            peW1t, pe_b1, peW2t, pe_b2, tmp, stats);
        ln_apply_kernel<<<2048, 256, 0, stream>>>(totE, tmp, stats, cntInvE, pe_g, pe_b, ec, en);

        float* t;
        t = hc; hc = hn; hn = t;
        t = ec; ec = en; en = t;
    }

    // ---- decoder -----------------------------------------------------------
    decoder_kernel<<<2048, 128, 0, stream>>>(
        N, hc, dec_W1, dec_b1, dec_W2, dec_b2, mean_ls, std_ls, outp);
}
